// MLPBlock_85048942396042
// MI455X (gfx1250) — compile-verified, matrix operand order checked by reading, not run on hardware
//
#include <hip/hip_runtime.h>
#include <math.h>

// ---------------------------------------------------------------------------
// MI455X (gfx1250) implementation.
//  prep_kernel   : pos-conv + argmax(layout) + bf16 pack of x into [pix][c]
//  wtrans_kernel : W_table f32 [l][c][o] -> bf16 [l][o][c] (B-operand layout)
//  wmma_kernel   : whole weight table resident in LDS (286KB < 320KB WGP LDS);
//                  per 16-pixel tile: for each of 35 labels compute
//                  D_l = A(16x64 bf16) x W_l(64x16) with chained
//                  v_wmma_f32_16x16x32_bf16, lane-select rows where arg==l.
//  norm_kernel   : per (b,o) instance-norm + leaky ReLU in place on d_out.
// ---------------------------------------------------------------------------

typedef __attribute__((ext_vector_type(16))) __bf16 v16bf;
typedef __attribute__((ext_vector_type(8)))  float  v8f;

#define Bn    2
#define Cc    64
#define Hh    128
#define Ww    128
#define HW    (Hh*Ww)          // 16384
#define NPIX  (Bn*HW)          // 32768
#define NLAB  35
#define FCO   64
#define EPSV  1e-5f
#define SLOPEV 0.01f

// workspace byte offsets
#define WT_BYTES  (NLAB*64u*64u*2u)       // 286720 bf16 weights
#define XBF_OFF   (WT_BYTES)              // 286720 (16B aligned)
#define XBF_BYTES (NPIX*(unsigned)Cc*2u)  // 4194304
#define ARG_OFF   (XBF_OFF + XBF_BYTES)

__device__ __forceinline__ unsigned short f2bf(float f) {
    unsigned int u = __float_as_uint(f);
    unsigned int r = (u + 0x7FFFu + ((u >> 16) & 1u)) >> 16;   // RNE
    return (unsigned short)r;
}

union V16 {
    v16bf v;
    struct { uint4 lo; uint4 hi; } q;
};

// ---------------------------------------------------------------------------
__global__ void prep_kernel(const float* __restrict__ x,
                            const float* __restrict__ layout,
                            const float* __restrict__ pr,
                            const float* __restrict__ conv_w,
                            const float* __restrict__ conv_b,
                            const float* __restrict__ pl,
                            unsigned short* __restrict__ xbf,
                            int* __restrict__ args)
{
    int p = blockIdx.x * blockDim.x + threadIdx.x;
    if (p >= NPIX) return;
    int b  = p >> 14;
    int hw = p & (HW - 1);
    int hh = hw >> 7;
    int ww = hw & (Ww - 1);

    // pos channels: [pa.x(h), pa.y(w), pl0, pl1, pr0, pr1, xcos, xsin, ycos, ysin]
    float px  = ((float)hh / (float)Hh) * 2.f - 1.f;
    float py  = ((float)ww / (float)Ww) * 2.f - 1.f;
    float pl0 = pl[hw];
    float pl1 = pl[HW + hw];
    float pr0 = pr[(b * 2 + 0) * HW + hw];
    float pr1 = pr[(b * 2 + 1) * HW + hw];
    const float K3 = 6.28318530717958647692f / 3.f;
    float ax = K3 * (float)(ww % 3);
    float ay = K3 * (float)(hh % 3);
    float s = conv_w[0] * px + conv_w[1] * py
            + conv_w[2] * pl0 + conv_w[3] * pl1
            + conv_w[4] * pr0 + conv_w[5] * pr1
            + conv_w[6] * __cosf(ax) + conv_w[7] * __sinf(ax)
            + conv_w[8] * __cosf(ay) + conv_w[9] * __sinf(ay)
            + conv_b[0];

    // argmax over labels (first occurrence wins)
    float best = layout[(b * NLAB) * HW + hw];
    int arg = 0;
    for (int n = 1; n < NLAB; ++n) {
        float v = layout[(b * NLAB + n) * HW + hw];
        if (v > best) { best = v; arg = n; }
    }
    args[p] = arg;

    // x + posconv, packed bf16 NHWC [pix][c]
    for (int c = 0; c < Cc; ++c) {
        float v = x[(b * Cc + c) * HW + hw] + s;
        xbf[p * Cc + c] = f2bf(v);
    }
}

// ---------------------------------------------------------------------------
__global__ void wtrans_kernel(const float* __restrict__ Wtab,
                              unsigned short* __restrict__ Wt)
{
    int i = blockIdx.x * blockDim.x + threadIdx.x;   // = l*4096 + o*64 + c
    if (i >= NLAB * 4096) return;
    int l   = i >> 12;
    int rem = i & 4095;
    int o   = rem >> 6;
    int c   = rem & 63;
    Wt[i] = f2bf(Wtab[l * 4096 + c * 64 + o]);       // [l][c][o] -> [l][o][c]
}

// ---------------------------------------------------------------------------
__global__ void __launch_bounds__(256, 1)
wmma_kernel(const unsigned short* __restrict__ WtG,
            const float* __restrict__ btab,
            const unsigned short* __restrict__ xbf,
            const int* __restrict__ args,
            float* __restrict__ out)
{
    extern __shared__ unsigned char smem[];
    __bf16* Wl    = (__bf16*)smem;                    // 35*64*64 bf16 = 286720 B
    float*  biasL = (float*)(smem + WT_BYTES);        // 35*64 f32     =  8960 B

    // cooperative LDS fill: whole weight table resident per WGP
    {
        const uint4* src = (const uint4*)WtG;
        uint4*       dst = (uint4*)smem;
        const int n16 = WT_BYTES / 16;                // 17920
        for (int i = threadIdx.x; i < n16; i += blockDim.x) dst[i] = src[i];
        for (int i = threadIdx.x; i < NLAB * FCO; i += blockDim.x) biasL[i] = btab[i];
    }
    __syncthreads();

    const int lane = threadIdx.x & 31;
    const int m    = lane & 15;       // A row / B col / D col within tile
    const int sel  = lane >> 4;       // lane half (per 16-bit WMMA VGPR layout)

    const int wavesPerBlk = blockDim.x >> 5;
    const int gwave  = blockIdx.x * wavesPerBlk + (threadIdx.x >> 5);
    const int nwaves = gridDim.x * wavesPerBlk;
    const int NTILES = NPIX / 16;     // 2048

    for (int t = gwave; t < NTILES; t += nwaves) {
        const int pixbase = t * 16;

        // per-lane row labels: D row M = r + sel*8
        int a[8];
        #pragma unroll
        for (int r = 0; r < 8; ++r) a[r] = args[pixbase + r + sel * 8];

        // A operands (K blocks 0..31 and 32..63), matching 16-bit A layout:
        // lane<16 -> K {kb*32+0..7, kb*32+16..23}; lane>=16 -> +8
        V16 A0, A1;
        {
            const unsigned short* base = xbf + (pixbase + m) * Cc;
            A0.q.lo = *(const uint4*)(base +  0 + sel * 8);
            A0.q.hi = *(const uint4*)(base + 16 + sel * 8);
            A1.q.lo = *(const uint4*)(base + 32 + sel * 8);
            A1.q.hi = *(const uint4*)(base + 48 + sel * 8);
        }
        // prefetch next tile's A rows
        __builtin_prefetch(xbf + (pixbase + 16 * nwaves + m) * Cc, 0, 0);

        float acc[4][8];
        #pragma unroll
        for (int nt = 0; nt < 4; ++nt)
            #pragma unroll
            for (int r = 0; r < 8; ++r) acc[nt][r] = 0.f;

        for (int l = 0; l < NLAB; ++l) {
            #pragma unroll
            for (int nt = 0; nt < 4; ++nt) {
                const __bf16* wb = Wl + (l << 12) + (nt * 16 + m) * 64; // row o, 64 ch
                V16 B0, B1;
                B0.q.lo = *(const uint4*)(wb +  0 + sel * 8);
                B0.q.hi = *(const uint4*)(wb + 16 + sel * 8);
                B1.q.lo = *(const uint4*)(wb + 32 + sel * 8);
                B1.q.hi = *(const uint4*)(wb + 48 + sel * 8);

                v8f d = {};
                d = __builtin_amdgcn_wmma_f32_16x16x32_bf16(
                        false, A0.v, false, B0.v, (short)0, d, false, false);
                d = __builtin_amdgcn_wmma_f32_16x16x32_bf16(
                        false, A1.v, false, B1.v, (short)0, d, false, false);

                #pragma unroll
                for (int r = 0; r < 8; ++r)
                    acc[nt][r] = (a[r] == l) ? d[r] : acc[nt][r];
            }
        }

        // bias + store pre-norm y into d_out in final [b][o][h][w] layout
        #pragma unroll
        for (int nt = 0; nt < 4; ++nt) {
            const int o = nt * 16 + m;
            #pragma unroll
            for (int r = 0; r < 8; ++r) {
                float y = acc[nt][r] + biasL[a[r] * FCO + o];
                int pix = pixbase + r + sel * 8;
                int b   = pix >> 14;
                int hw  = pix & (HW - 1);
                out[(b * FCO + o) * HW + hw] = y;
            }
        }
    }
}

// ---------------------------------------------------------------------------
__global__ void norm_kernel(float* __restrict__ out)
{
    __shared__ float ssum[256];
    __shared__ float ssq[256];
    const int ch = blockIdx.x;                 // b*64 + o, 128 channels
    float* base = out + (size_t)ch * HW;

    float s = 0.f, q = 0.f;
    for (int i = threadIdx.x; i < HW; i += blockDim.x) {
        float v = base[i];
        s += v; q += v * v;
    }
    ssum[threadIdx.x] = s; ssq[threadIdx.x] = q;
    __syncthreads();
    for (int off = 128; off > 0; off >>= 1) {
        if ((int)threadIdx.x < off) {
            ssum[threadIdx.x] += ssum[threadIdx.x + off];
            ssq[threadIdx.x]  += ssq[threadIdx.x + off];
        }
        __syncthreads();
    }
    const float inv = 1.f / (float)HW;
    float mu  = ssum[0] * inv;
    float var = ssq[0] * inv - mu * mu;
    float rs  = rsqrtf(var + EPSV);
    for (int i = threadIdx.x; i < HW; i += blockDim.x) {
        float v = (base[i] - mu) * rs;
        base[i] = (v >= 0.f) ? v : SLOPEV * v;
    }
}

// ---------------------------------------------------------------------------
extern "C" void kernel_launch(void* const* d_in, const int* in_sizes, int n_in,
                              void* d_out, int out_size, void* d_ws, size_t ws_size,
                              hipStream_t stream)
{
    const float* x      = (const float*)d_in[0];
    const float* layout = (const float*)d_in[1];
    const float* pr     = (const float*)d_in[2];
    const float* conv_w = (const float*)d_in[3];
    const float* conv_b = (const float*)d_in[4];
    const float* Wtab   = (const float*)d_in[5];
    const float* btab   = (const float*)d_in[6];
    const float* pl     = (const float*)d_in[7];
    float* out          = (float*)d_out;

    unsigned char* ws = (unsigned char*)d_ws;
    unsigned short* Wt  = (unsigned short*)(ws);            // bf16 [l][o][c]
    unsigned short* xbf = (unsigned short*)(ws + XBF_OFF);  // bf16 [pix][c]
    int*            arg = (int*)(ws + ARG_OFF);             // int  [pix]

    prep_kernel<<<NPIX / 256, 256, 0, stream>>>(x, layout, pr, conv_w, conv_b, pl, xbf, arg);
    wtrans_kernel<<<(NLAB * 4096 + 255) / 256, 256, 0, stream>>>(Wtab, Wt);

    const size_t shbytes = (size_t)WT_BYTES + (size_t)NLAB * FCO * sizeof(float); // 295680
    hipFuncSetAttribute((const void*)wmma_kernel,
                        hipFuncAttributeMaxDynamicSharedMemorySize, (int)shbytes);
    wmma_kernel<<<128, 256, shbytes, stream>>>(Wt, btab, xbf, arg, out);

    norm_kernel<<<Bn * FCO, 256, 0, stream>>>(out);
}
